// ReportDecoder_34067680591929
// MI455X (gfx1250) — matrix operand order repaired and checked by reading, not compiled, not run on hardware
//
#include <hip/hip_runtime.h>
#include <hip/hip_bf16.h>

// ---------------------------------------------------------------------------
// ReportDecoder (LSTM decoder) for MI455X / gfx1250, wave32 + WMMA f16->f32.
//
// Algebraic restructuring:
//   x_t = [image_mean | view | begin_t | lab_t]  (D_IN = 1105)
//   gates = x_t @ W_ih^T + b_ih + h @ W_hh^T + b_hh
// The [image_mean|view] part (1040 cols) is time-invariant:
//   BASE  = [img_mean|view] @ W_ih[:, :1040]^T + b_ih + b_hh      (once)
//   gates = BASE + (t==0)*w_begin + [h | lab_t] @ [W_hh | W_lab]^T  (K=1088)
// ---------------------------------------------------------------------------

#define B_     512
#define S_     49
#define E_     1024
#define H_     1024
#define V2_    16
#define L_     64
#define T_     48
#define DIN_   1105
#define K2_    1088   // H + L (per-step GEMM reduction dim)
#define K0_    1056   // E + V2 padded to multiple of 32
#define N4_    4096   // 4*H gate width
#define NOUTP_ 1152   // L+H+2 = 1090 padded to multiple of 128
#define NOUT_  1090

typedef __attribute__((ext_vector_type(16))) _Float16 v16h;
typedef __attribute__((ext_vector_type(8)))  float    v8f;

static constexpr size_t OFF_TOPIC = (size_t)T_ * B_ * L_;
static constexpr size_t OFF_STOP  = OFF_TOPIC + (size_t)T_ * B_ * H_;
static constexpr size_t OFF_TEMP  = OFF_STOP + (size_t)T_ * B_;

union FragU { struct { float4 lo, hi; } q; v16h v; };

__device__ __forceinline__ v8f zero8() {
  v8f z;
#pragma unroll
  for (int i = 0; i < 8; ++i) z[i] = 0.0f;
  return z;
}

// A fragment 16x32 f16 (M x K), row-major source with leading dim ld.
// ISA layout: lanes 0-15 row M=lane, K {k0..k0+7, k0+16..k0+23};
//             lanes 16-31 row M=lane-16, K {k0+8..k0+15, k0+24..k0+31}.
__device__ __forceinline__ v16h load_a_frag(const _Float16* __restrict__ X,
                                            int ld, int row0, int k0) {
  const int lane = threadIdx.x & 31;
  const _Float16* p = X + (size_t)(row0 + (lane & 15)) * ld + k0 + ((lane & 16) ? 8 : 0);
  FragU u;
  u.q.lo = *reinterpret_cast<const float4*>(p);
  u.q.hi = *reinterpret_cast<const float4*>(p + 16);
  return u.v;
}

// B fragment 32x16 f16 (K x N). Source stored as (N,K) row-major (= W of y=xW^T).
// ISA layout: lanes 0-15 col N=lane, K {k0..k0+7, k0+16..k0+23};
//             lanes 16-31 col N=lane-16, K {k0+8..k0+15, k0+24..k0+31}.
__device__ __forceinline__ v16h load_b_frag(const _Float16* __restrict__ W,
                                            int ld, int n0, int k0) {
  const int lane = threadIdx.x & 31;
  const _Float16* p = W + (size_t)(n0 + (lane & 15)) * ld + k0 + ((lane & 16) ? 8 : 0);
  FragU u;
  u.q.lo = *reinterpret_cast<const float4*>(p);
  u.q.hi = *reinterpret_cast<const float4*>(p + 16);
  return u.v;
}

__device__ __forceinline__ v8f wmma16(v16h a, v16h b, v8f c) {
  return __builtin_amdgcn_wmma_f32_16x16x32_f16(false, a, false, b, (short)0, c,
                                                false, false);
}

__device__ __forceinline__ float sigmoidf_(float x) { return 1.0f / (1.0f + __expf(-x)); }

// ---------------------------------------------------------------------------
// Prep kernels (cheap elementwise / reduction)
// ---------------------------------------------------------------------------

// X0 (B x 1056) f16: [image_mean | view | zero-pad]
__global__ void k_prep_x0(const float* __restrict__ img,
                          const float* __restrict__ view,
                          _Float16* __restrict__ X0) {
  int i = blockIdx.x * 256 + threadIdx.x;
  if (i >= B_ * K0_) return;
  int b = i / K0_, c = i % K0_;
  float v;
  if (c < E_) {
    const float* p = img + (size_t)b * S_ * E_ + c;
    float s = 0.0f;
    for (int j = 0; j < S_; ++j) s += p[(size_t)j * E_];
    v = s * (1.0f / (float)S_);
  } else if (c < E_ + V2_) {
    v = view[b * V2_ + (c - E_)];
  } else {
    v = 0.0f;
  }
  X0[i] = (_Float16)v;
}

// Wbase (4096 x 1056) f16 from w_ih[:, :1040], zero-padded K
__global__ void k_pack_wbase(const float* __restrict__ w_ih, _Float16* __restrict__ W) {
  int i = blockIdx.x * 256 + threadIdx.x;
  if (i >= N4_ * K0_) return;
  int n = i / K0_, k = i % K0_;
  float v = (k < E_ + V2_) ? w_ih[(size_t)n * DIN_ + k] : 0.0f;
  W[i] = (_Float16)v;
}

// W2 (4096 x 1088) f16 = [w_hh | w_ih[:, 1041:1105]]
__global__ void k_pack_w2(const float* __restrict__ w_hh,
                          const float* __restrict__ w_ih,
                          _Float16* __restrict__ W) {
  int i = blockIdx.x * 256 + threadIdx.x;
  if (i >= N4_ * K2_) return;
  int n = i / K2_, k = i % K2_;
  float v = (k < H_) ? w_hh[(size_t)n * H_ + k]
                     : w_ih[(size_t)n * DIN_ + (E_ + V2_ + 1) + (k - H_)];
  W[i] = (_Float16)v;
}

__global__ void k_cvt_f16(const float* __restrict__ in, _Float16* __restrict__ out, int n) {
  int i = blockIdx.x * 256 + threadIdx.x;
  if (i < n) out[i] = (_Float16)in[i];
}

// Wout (1152 x 1024) f16 from fc_w (1090 x 1024), zero-padded rows
__global__ void k_pack_wout(const float* __restrict__ fc_w, _Float16* __restrict__ W) {
  int i = blockIdx.x * 256 + threadIdx.x;
  if (i >= NOUTP_ * H_) return;
  int n = i >> 10, k = i & (H_ - 1);
  float v = (n < NOUT_) ? fc_w[(size_t)n * H_ + k] : 0.0f;
  W[i] = (_Float16)v;
}

// labx (T x B x 64) f16: labx[0]=0, labx[t]=label[:, t-1]
__global__ void k_pack_labx(const float* __restrict__ label, _Float16* __restrict__ labx) {
  int i = blockIdx.x * 256 + threadIdx.x;
  if (i >= T_ * B_ * L_) return;
  int l = i & (L_ - 1);
  int b = (i >> 6) & (B_ - 1);
  int t = i >> 15;  // 512*64 = 2^15
  float v = (t == 0) ? 0.0f : label[(size_t)b * (T_ + 1) * L_ + (size_t)(t - 1) * L_ + l];
  labx[i] = (_Float16)v;
}

// ---------------------------------------------------------------------------
// WMMA GEMM kernels. Block = 256 threads = 8 waves (2 M x 4 N).
// Wave computes a 32x32 output tile per stream; block tile M=64, N=128.
// ---------------------------------------------------------------------------

// h0/m0: tanh(X0[:, :1024] @ {fc_h_w,fc_m_w}^T + bias); write h f16 + m f32
__global__ __launch_bounds__(256) void k_init_h0m0(
    const _Float16* __restrict__ X0,
    const _Float16* __restrict__ Wh, const _Float16* __restrict__ Wm,
    const float* __restrict__ bh, const float* __restrict__ bm,
    _Float16* __restrict__ Xh, float* __restrict__ mbuf) {
  const int w = threadIdx.x >> 5;
  const int row0 = blockIdx.x * 64 + (w >> 2) * 32;
  const int col0 = blockIdx.y * 128 + (w & 3) * 32;
  v8f acc[2][2][2];
#pragma unroll
  for (int s = 0; s < 2; ++s)
#pragma unroll
    for (int a = 0; a < 2; ++a)
#pragma unroll
      for (int b = 0; b < 2; ++b) acc[s][a][b] = zero8();

  for (int kt = 0; kt < 32; ++kt) {
    const int k0 = kt * 32;
    v16h a0 = load_a_frag(X0, K0_, row0, k0);
    v16h a1 = load_a_frag(X0, K0_, row0 + 16, k0);
#pragma unroll
    for (int nn = 0; nn < 2; ++nn) {
      v16h bh16 = load_b_frag(Wh, H_, col0 + nn * 16, k0);
      acc[0][0][nn] = wmma16(a0, bh16, acc[0][0][nn]);
      acc[0][1][nn] = wmma16(a1, bh16, acc[0][1][nn]);
      v16h bm16 = load_b_frag(Wm, H_, col0 + nn * 16, k0);
      acc[1][0][nn] = wmma16(a0, bm16, acc[1][0][nn]);
      acc[1][1][nn] = wmma16(a1, bm16, acc[1][1][nn]);
    }
  }
  const int lane = threadIdx.x & 31;
  const int rhi = (lane & 16) ? 8 : 0;
  const int cl = lane & 15;
#pragma unroll
  for (int mm = 0; mm < 2; ++mm)
#pragma unroll
    for (int nn = 0; nn < 2; ++nn) {
      const int c = col0 + nn * 16 + cl;
#pragma unroll
      for (int r = 0; r < 8; ++r) {
        const int b = row0 + mm * 16 + rhi + r;
        const size_t bc = (size_t)b * H_ + c;
        Xh[bc] = (_Float16)tanhf(acc[0][mm][nn][r] + bh[c]);
        mbuf[bc] = tanhf(acc[1][mm][nn][r] + bm[c]);
      }
    }
}

// BASE = X0 @ Wbase^T + b_ih + b_hh   (B x 4096 f32)
__global__ __launch_bounds__(256) void k_base(
    const _Float16* __restrict__ X0, const _Float16* __restrict__ Wb,
    const float* __restrict__ b_ih, const float* __restrict__ b_hh,
    float* __restrict__ BASE) {
  const int w = threadIdx.x >> 5;
  const int row0 = blockIdx.x * 64 + (w >> 2) * 32;
  const int col0 = blockIdx.y * 128 + (w & 3) * 32;
  v8f acc[2][2];
#pragma unroll
  for (int a = 0; a < 2; ++a)
#pragma unroll
    for (int b = 0; b < 2; ++b) acc[a][b] = zero8();

  for (int kt = 0; kt < 33; ++kt) {
    const int k0 = kt * 32;
    v16h a0 = load_a_frag(X0, K0_, row0, k0);
    v16h a1 = load_a_frag(X0, K0_, row0 + 16, k0);
#pragma unroll
    for (int nn = 0; nn < 2; ++nn) {
      v16h bf = load_b_frag(Wb, K0_, col0 + nn * 16, k0);
      acc[0][nn] = wmma16(a0, bf, acc[0][nn]);
      acc[1][nn] = wmma16(a1, bf, acc[1][nn]);
    }
  }
  const int lane = threadIdx.x & 31;
  const int rhi = (lane & 16) ? 8 : 0;
  const int cl = lane & 15;
#pragma unroll
  for (int mm = 0; mm < 2; ++mm)
#pragma unroll
    for (int nn = 0; nn < 2; ++nn) {
      const int c = col0 + nn * 16 + cl;
      const float bias = b_ih[c] + b_hh[c];
#pragma unroll
      for (int r = 0; r < 8; ++r) {
        const int b = row0 + mm * 16 + rhi + r;
        BASE[(size_t)b * N4_ + c] = acc[mm][nn][r] + bias;
      }
    }
}

// gates + LSTM cell: gates = BASE + beg + [h|lab_t] @ W2^T ; update m, emit h f16
__global__ __launch_bounds__(256) void k_gates(
    const _Float16* __restrict__ Xh,    // h_{t-1}, (B x 1024) f16
    const _Float16* __restrict__ labT,  // lab_t,   (B x 64) f16
    const _Float16* __restrict__ W2,    // (4096 x 1088) f16
    const float* __restrict__ BASE,     // (B x 4096)
    const float* __restrict__ w_ih,     // raw, for begin column 1040
    float* __restrict__ mbuf,           // (B x 1024), in-place
    _Float16* __restrict__ Xout,        // h_t, (B x 1024) f16
    int t) {
  const int w = threadIdx.x >> 5;
  const int row0 = blockIdx.x * 64 + (w >> 2) * 32;
  const int col0 = blockIdx.y * 128 + (w & 3) * 32;
  v8f acc[4][2][2];
#pragma unroll
  for (int g = 0; g < 4; ++g)
#pragma unroll
    for (int a = 0; a < 2; ++a)
#pragma unroll
      for (int b = 0; b < 2; ++b) acc[g][a][b] = zero8();

  for (int kt = 0; kt < 34; ++kt) {
    const int k0 = kt * 32;
    v16h a0, a1;
    if (kt < 32) {
      a0 = load_a_frag(Xh, H_, row0, k0);
      a1 = load_a_frag(Xh, H_, row0 + 16, k0);
    } else {
      a0 = load_a_frag(labT, L_, row0, k0 - H_);
      a1 = load_a_frag(labT, L_, row0 + 16, k0 - H_);
    }
#pragma unroll
    for (int g = 0; g < 4; ++g)
#pragma unroll
      for (int nn = 0; nn < 2; ++nn) {
        v16h bf = load_b_frag(W2, K2_, g * H_ + col0 + nn * 16, k0);
        acc[g][0][nn] = wmma16(a0, bf, acc[g][0][nn]);
        acc[g][1][nn] = wmma16(a1, bf, acc[g][1][nn]);
      }
  }

  const bool isT0 = (t == 0);
  const int lane = threadIdx.x & 31;
  const int rhi = (lane & 16) ? 8 : 0;
  const int cl = lane & 15;
#pragma unroll
  for (int mm = 0; mm < 2; ++mm)
#pragma unroll
    for (int nn = 0; nn < 2; ++nn) {
      const int c = col0 + nn * 16 + cl;
      float bg[4];
#pragma unroll
      for (int g = 0; g < 4; ++g)
        bg[g] = isT0 ? w_ih[(size_t)(g * H_ + c) * DIN_ + (E_ + V2_)] : 0.0f;
#pragma unroll
      for (int r = 0; r < 8; ++r) {
        const int b = row0 + mm * 16 + rhi + r;
        const size_t bofs = (size_t)b * N4_;
        const float vi = acc[0][mm][nn][r] + BASE[bofs + c] + bg[0];
        const float vf = acc[1][mm][nn][r] + BASE[bofs + H_ + c] + bg[1];
        const float vg = acc[2][mm][nn][r] + BASE[bofs + 2 * H_ + c] + bg[2];
        const float vo = acc[3][mm][nn][r] + BASE[bofs + 3 * H_ + c] + bg[3];
        const size_t bc = (size_t)b * H_ + c;
        const float mo = mbuf[bc];
        const float mn = sigmoidf_(vf) * mo + sigmoidf_(vi) * tanhf(vg);
        mbuf[bc] = mn;
        Xout[bc] = (_Float16)(sigmoidf_(vo) * tanhf(mn));
      }
    }
}

// out = h_t @ fc_w^T + fc_b, fused sigmoid/relu/exp epilogue with length mask
__global__ __launch_bounds__(256) void k_out(
    const _Float16* __restrict__ Xh,   // h_t f16
    const _Float16* __restrict__ Wout, // (1152 x 1024) f16 (rows>=1090 zero)
    const float* __restrict__ fc_b,
    const int* __restrict__ length,
    float* __restrict__ out, int t) {
  const int w = threadIdx.x >> 5;
  const int row0 = blockIdx.x * 64 + (w >> 2) * 32;
  const int col0 = blockIdx.y * 128 + (w & 3) * 32;
  v8f acc[2][2];
#pragma unroll
  for (int a = 0; a < 2; ++a)
#pragma unroll
    for (int b = 0; b < 2; ++b) acc[a][b] = zero8();

  for (int kt = 0; kt < 32; ++kt) {
    const int k0 = kt * 32;
    v16h a0 = load_a_frag(Xh, H_, row0, k0);
    v16h a1 = load_a_frag(Xh, H_, row0 + 16, k0);
#pragma unroll
    for (int nn = 0; nn < 2; ++nn) {
      v16h bf = load_b_frag(Wout, H_, col0 + nn * 16, k0);
      acc[0][nn] = wmma16(a0, bf, acc[0][nn]);
      acc[1][nn] = wmma16(a1, bf, acc[1][nn]);
    }
  }
  const int lane = threadIdx.x & 31;
  const int rhi = (lane & 16) ? 8 : 0;
  const int cl = lane & 15;
#pragma unroll
  for (int mm = 0; mm < 2; ++mm)
#pragma unroll
    for (int nn = 0; nn < 2; ++nn) {
      const int n = col0 + nn * 16 + cl;
      if (n >= NOUT_) continue;
      const float bias = fc_b[n];
#pragma unroll
      for (int r = 0; r < 8; ++r) {
        const int b = row0 + mm * 16 + rhi + r;
        const float msk = (t < length[b]) ? 1.0f : 0.0f;
        const float v = acc[mm][nn][r] + bias;
        if (n < L_)
          out[(size_t)t * B_ * L_ + (size_t)b * L_ + n] = msk * sigmoidf_(v);
        else if (n < L_ + H_)
          out[OFF_TOPIC + (size_t)t * B_ * H_ + (size_t)b * H_ + (n - L_)] =
              msk * fmaxf(v, 0.0f);
        else if (n == L_ + H_)
          out[OFF_STOP + (size_t)t * B_ + b] = msk * sigmoidf_(v);
        else
          out[OFF_TEMP + (size_t)t * B_ + b] = msk * __expf(v);
      }
    }
}

// ---------------------------------------------------------------------------

static inline size_t align256(size_t x) { return (x + 255) & ~(size_t)255; }

extern "C" void kernel_launch(void* const* d_in, const int* in_sizes, int n_in,
                              void* d_out, int out_size, void* d_ws, size_t ws_size,
                              hipStream_t stream) {
  (void)in_sizes; (void)n_in; (void)out_size; (void)ws_size;
  const float* image  = (const float*)d_in[0];
  const float* view   = (const float*)d_in[1];
  const float* label  = (const float*)d_in[2];
  const int*   length = (const int*)d_in[3];
  const float* fc_h_w = (const float*)d_in[4];
  const float* fc_h_b = (const float*)d_in[5];
  const float* fc_m_w = (const float*)d_in[6];
  const float* fc_m_b = (const float*)d_in[7];
  const float* w_ih   = (const float*)d_in[8];
  const float* w_hh   = (const float*)d_in[9];
  const float* b_ih   = (const float*)d_in[10];
  const float* b_hh   = (const float*)d_in[11];
  const float* fc_w   = (const float*)d_in[12];
  const float* fc_b   = (const float*)d_in[13];
  float* out = (float*)d_out;
  char* ws = (char*)d_ws;

  size_t off = 0;
  _Float16* X0    = (_Float16*)(ws + off); off = align256(off + (size_t)B_ * K0_ * 2);
  _Float16* Wbase = (_Float16*)(ws + off); off = align256(off + (size_t)N4_ * K0_ * 2);
  _Float16* W2    = (_Float16*)(ws + off); off = align256(off + (size_t)N4_ * K2_ * 2);
  _Float16* Wh16  = (_Float16*)(ws + off); off = align256(off + (size_t)H_ * E_ * 2);
  _Float16* Wm16  = (_Float16*)(ws + off); off = align256(off + (size_t)H_ * E_ * 2);
  _Float16* Wout  = (_Float16*)(ws + off); off = align256(off + (size_t)NOUTP_ * H_ * 2);
  _Float16* labx  = (_Float16*)(ws + off); off = align256(off + (size_t)T_ * B_ * L_ * 2);
  float*    BASE  = (float*)(ws + off);    off = align256(off + (size_t)B_ * N4_ * 4);
  float*    mbuf  = (float*)(ws + off);    off = align256(off + (size_t)B_ * H_ * 4);
  _Float16* XhA   = (_Float16*)(ws + off); off = align256(off + (size_t)B_ * H_ * 2);
  _Float16* XhB   = (_Float16*)(ws + off); off = align256(off + (size_t)B_ * H_ * 2);

  auto blocks = [](size_t n) { return (unsigned)((n + 255) / 256); };

  // ---- one-time prep (re-run every call; deterministic) ----
  k_prep_x0<<<blocks((size_t)B_ * K0_), 256, 0, stream>>>(image, view, X0);
  k_pack_wbase<<<blocks((size_t)N4_ * K0_), 256, 0, stream>>>(w_ih, Wbase);
  k_pack_w2<<<blocks((size_t)N4_ * K2_), 256, 0, stream>>>(w_hh, w_ih, W2);
  k_cvt_f16<<<blocks((size_t)H_ * E_), 256, 0, stream>>>(fc_h_w, Wh16, H_ * E_);
  k_cvt_f16<<<blocks((size_t)H_ * E_), 256, 0, stream>>>(fc_m_w, Wm16, H_ * E_);
  k_pack_wout<<<blocks((size_t)NOUTP_ * H_), 256, 0, stream>>>(fc_w, Wout);
  k_pack_labx<<<blocks((size_t)T_ * B_ * L_), 256, 0, stream>>>(label, labx);

  // ---- init GEMMs ----
  k_init_h0m0<<<dim3(B_ / 64, H_ / 128), 256, 0, stream>>>(X0, Wh16, Wm16,
                                                           fc_h_b, fc_m_b, XhA, mbuf);
  k_base<<<dim3(B_ / 64, N4_ / 128), 256, 0, stream>>>(X0, Wbase, b_ih, b_hh, BASE);

  // ---- sequential LSTM steps (double-buffered h) ----
  for (int t = 0; t < T_; ++t) {
    const _Float16* hin = (t & 1) ? XhB : XhA;
    _Float16* hout = (t & 1) ? XhA : XhB;
    k_gates<<<dim3(B_ / 64, H_ / 128), 256, 0, stream>>>(
        hin, labx + (size_t)t * B_ * L_, W2, BASE, w_ih, mbuf, hout, t);
    k_out<<<dim3(B_ / 64, NOUTP_ / 128), 256, 0, stream>>>(
        hout, Wout, fc_b, length, out, t);
  }
}
